// TemporalGAT_38500086841700
// MI455X (gfx1250) — matrix operand order repaired and checked by reading, not compiled, not run on hardware
//
#include <hip/hip_runtime.h>

// ---------------------------------------------------------------------------
// TemporalGAT for MI455X (gfx1250, wave32).
//   - layer-1 projection  : v_wmma_f32_16x16x4_f32   (IN=4 is the exact shape)
//   - layer-2 projection  : v_wmma_f32_16x16x32_bf16 (bf16 in, f32 accum)
//   - W2 panel staged to LDS via TDM tensor_load_to_lds (+ s_wait_tensorcnt)
//   - segment softmax / aggregation: coalesced f32 global atomics
// ---------------------------------------------------------------------------

typedef __bf16 bf16_t;
typedef __attribute__((ext_vector_type(2)))  float  v2f;
typedef __attribute__((ext_vector_type(8)))  float  v8f;
typedef __attribute__((ext_vector_type(8)))  bf16_t v8bf;
typedef __attribute__((ext_vector_type(16))) bf16_t v16bf;
typedef __attribute__((ext_vector_type(4)))  unsigned int u32x4;
typedef __attribute__((ext_vector_type(4)))  int          i32x4;
typedef __attribute__((ext_vector_type(8)))  int          i32x8;

#if __has_builtin(__builtin_amdgcn_tensor_load_to_lds) && __has_builtin(__builtin_amdgcn_s_wait_tensorcnt)
#define HAVE_TDM 1
#else
#define HAVE_TDM 0
#endif

#define HID   64
#define HEADS 4
#define NGR   64
#define NEG_SLOPE 0.2f
#define W2SWZ_ELEMS (4 * 8 * 32 * 16)   // 16384 bf16 = 32 KB = 8192 dwords

__device__ __forceinline__ float lrelu(float v) { return v > 0.f ? v : NEG_SLOPE * v; }
__device__ __forceinline__ float elu(float v)   { return v > 0.f ? v : (expf(v) - 1.f); }

__device__ __forceinline__ void atomicMaxF(float* addr, float v) {
  if (v >= 0.f) atomicMax((int*)addr, __float_as_int(v));
  else          atomicMin((unsigned int*)addr, __float_as_uint(v));
}

// ----------------------------- utility ------------------------------------
__global__ void k_zero(float* __restrict__ p, long n) {
  long i = (long)blockIdx.x * blockDim.x + threadIdx.x;
  if (i < n) p[i] = 0.f;
}

// ------------------- layer 1: h1 = x @ W1  (WMMA 16x16x4 f32) --------------
// x: [N,4]  W1: [4,256]  h1: [N,256]
__global__ void k_gemm1(const float* __restrict__ x, const float* __restrict__ W1,
                        float* __restrict__ h1, int n) {
  const int wave = threadIdx.x >> 5;      // 8 waves
  const int lane = threadIdx.x & 31;
  const int m    = lane & 15;
  const int r0   = blockIdx.x * 16;
  int r = r0 + m; if (r >= n) r = n - 1;  // clamped load, masked store
  const int khalf = (lane < 16) ? 0 : 2;  // 32-bit A 16x4 layout (ISA 7.12.2)
  const bool full = (r0 + 16) <= n;       // uniform fast path
#if __has_builtin(__builtin_amdgcn_wmma_f32_16x16x4_f32)
  v2f a;
  a.x = x[r * 4 + khalf + 0];
  a.y = x[r * 4 + khalf + 1];
  #pragma unroll
  for (int t = 0; t < 2; ++t) {
    const int ct  = wave * 2 + t;         // 16 column tiles, 8 waves x 2
    const int col = ct * 16 + m;
    v2f b;                                 // 32-bit B 4x16: low lanes K0..1, high K2..3
    b.x = W1[(khalf + 0) * 256 + col];
    b.y = W1[(khalf + 1) * 256 + col];
    v8f c = {};
    c = __builtin_amdgcn_wmma_f32_16x16x4_f32(false, a, false, b, (short)0, c, false, false);
    const int rbase = r0 + ((lane < 16) ? 0 : 8);
    if (full) {
      #pragma unroll
      for (int v = 0; v < 8; ++v) h1[(long)(rbase + v) * 256 + col] = c[v];
    } else {
      #pragma unroll
      for (int v = 0; v < 8; ++v)
        if (rbase + v < n) h1[(long)(rbase + v) * 256 + col] = c[v];
    }
  }
#else
  for (int t = 0; t < 2; ++t) {
    const int ct = wave * 2 + t;
    const int col = ct * 16 + m;
    for (int v = 0; v < 8; ++v) {
      int row = r0 + v + ((lane < 16) ? 0 : 8);
      if (row < n) {
        float acc = 0.f;
        for (int k = 0; k < 4; ++k) acc += x[row * 4 + k] * W1[k * 256 + col];
        h1[(long)row * 256 + col] = acc;
      }
    }
  }
#endif
}

// alpha_src/alpha_dst for layer 1: one wave per node, shfl-xor reduction
__global__ void k_alpha1(const float* __restrict__ h1, const float* __restrict__ a_src,
                         const float* __restrict__ a_dst, float* __restrict__ as,
                         float* __restrict__ ad, int n) {
  const int wave = threadIdx.x >> 5;   // 8 waves/block
  const int lane = threadIdx.x & 31;
  long node = (long)blockIdx.x * 8 + wave;
  if (node >= n) return;               // wave-uniform exit
  const float* hp  = h1 + node * 256 + lane * 8;   // lane covers flat [h*64+c] chunk
  const float* asp = a_src + lane * 8;
  const float* adp = a_dst + lane * 8;
  float s = 0.f, d = 0.f;
  #pragma unroll
  for (int k = 0; k < 8; ++k) { float v = hp[k]; s += v * asp[k]; d += v * adp[k]; }
  // reduce within groups of 8 lanes (one head per group)
  s += __shfl_xor(s, 1); d += __shfl_xor(d, 1);
  s += __shfl_xor(s, 2); d += __shfl_xor(d, 2);
  s += __shfl_xor(s, 4); d += __shfl_xor(d, 4);
  if ((lane & 7) == 0) {
    int h = lane >> 3;
    as[node * HEADS + h] = s;
    ad[node * HEADS + h] = d;
  }
}

// init running max with the self-loop edge value
__global__ void k_mxinit1(const float* __restrict__ as, const float* __restrict__ ad,
                          float* __restrict__ mx, int n) {
  long i = (long)blockIdx.x * blockDim.x + threadIdx.x;
  if (i >= (long)n * HEADS) return;
  mx[i] = lrelu(as[i] + ad[i]);
}

__global__ void k_edgemax1(const int* __restrict__ ei, const float* __restrict__ as,
                           const float* __restrict__ ad, float* __restrict__ mx, int ne) {
  long i = (long)blockIdx.x * blockDim.x + threadIdx.x;
  if (i >= (long)ne * HEADS) return;
  int e = (int)(i >> 2), h = (int)(i & 3);
  int s = ei[e], d = ei[ne + e];
  atomicMaxF(&mx[(long)d * HEADS + h], lrelu(as[(long)s * HEADS + h] + ad[(long)d * HEADS + h]));
}

// channel-parallel messages: thread i -> (pair i>>6, channel i&63); coalesced atomics
__global__ void k_msg1(const int* __restrict__ ei, const float* __restrict__ h1,
                       const float* __restrict__ as, const float* __restrict__ ad,
                       const float* __restrict__ mx, float* __restrict__ den,
                       float* __restrict__ oacc, int ne, int n) {
  long i = (long)blockIdx.x * blockDim.x + threadIdx.x;
  long total = (long)(ne + n) * HEADS * HID;
  if (i >= total) return;
  int  c  = (int)(i & 63);
  long ph = i >> 6;
  int  h  = (int)(ph & 3);
  int  p  = (int)(ph >> 2);
  int s, d;
  if (p < ne) { s = ei[p]; d = ei[ne + p]; } else { s = d = p - ne; }
  float e  = lrelu(as[(long)s * HEADS + h] + ad[(long)d * HEADS + h]);
  float ex = expf(e - mx[(long)d * HEADS + h]);
  if (c == 0) atomicAdd(&den[(long)d * HEADS + h], ex);
  atomicAdd(&oacc[(long)d * 256 + h * HID + c], ex * h1[(long)s * 256 + h * HID + c]);
}

// finalize layer 1: ELU((acc/den)+b1) -> bf16 (input of layer-2 WMMA)
__global__ void k_fin1(const float* __restrict__ oacc, const float* __restrict__ den,
                       const float* __restrict__ b1, bf16_t* __restrict__ obf, int n) {
  long i = (long)blockIdx.x * blockDim.x + threadIdx.x;
  if (i >= (long)n * 256) return;
  int node = (int)(i >> 8), j = (int)(i & 255);
  float v = oacc[i] / den[(long)node * HEADS + (j >> 6)] + b1[j];
  obf[i] = (bf16_t)elu(v);
}

// pre-swizzle W2 [256,64] f32 -> bf16 in exact B-fragment lane layout:
// [ctile(4)][kstep(8)][lane(32)][e(16)],  K = kstep*32 + (lane<16?0:16) + e,
// Ncol = ctile*16 + (lane&15)
__global__ void k_prepw2(const float* __restrict__ W2, bf16_t* __restrict__ Bswz) {
  int idx = blockIdx.x * blockDim.x + threadIdx.x;
  if (idx >= W2SWZ_ELEMS) return;
  int e    = idx & 15;
  int lane = (idx >> 4) & 31;
  int ks   = (idx >> 9) & 7;
  int ct   = idx >> 12;
  int col  = ct * 16 + (lane & 15);
  int k    = ks * 32 + ((lane < 16) ? 0 : 16) + e;
  Bswz[idx] = (bf16_t)W2[k * 64 + col];
}

// ------------------- layer 2: h2 = o1 @ W2  (WMMA bf16 16x16x32) -----------
// A: [N,256] bf16 row-major, Bswz pre-laid-out; W2 panel staged to LDS by TDM.
__global__ void k_gemm2(const bf16_t* __restrict__ A, const bf16_t* __restrict__ Bswz,
                        float* __restrict__ h2, int n) {
  const int wave = threadIdx.x >> 5;     // 4 waves -> 4 column tiles of 16
  const int lane = threadIdx.x & 31;
  const int m    = lane & 15;
  const int r0   = blockIdx.x * 16;
  int r = r0 + m; if (r >= n) r = n - 1;
  const int  base = (lane < 16) ? 0 : 16; // contiguous K order: K = 16*half + e (matches Bswz)
  const bool full = (r0 + 16) <= n;

#if HAVE_TDM
  __shared__ __align__(128) bf16_t sB[W2SWZ_ELEMS];   // 32 KB
  if (threadIdx.x < 32) {                // wave 0 issues the TDM (EXEC ignored)
    unsigned lds = (unsigned)(unsigned long long)
        (__attribute__((address_space(3))) bf16_t*)sB;
    unsigned long long ga = (unsigned long long)Bswz;
    // D# per CDNA5 ISA: 1-D tile of 8192 dwords.
    u32x4 g0;
    g0[0] = 1u;                                             // count=1 (user)
    g0[1] = lds;                                            // lds_addr (bytes)
    g0[2] = (unsigned)(ga & 0xffffffffu);                   // global_addr[31:0]
    g0[3] = (unsigned)((ga >> 32) & 0x01ffffffu) | (2u << 30); // addr[56:32]|type=2
    i32x8 g1;
    g1[0] = 0x00020000;            // data_size=2 -> 4-byte elements
    g1[1] = (int)(0x2000u << 16);  // tensor_dim0 = 8192 (lo16)
    g1[2] = (int)(1u  << 16);      // tensor_dim0 hi=0 | tensor_dim1 = 1 (lo16)
    g1[3] = (int)(0x2000u << 16);  // tensor_dim1 hi=0 | tile_dim0 = 8192
    g1[4] = 0;                     // tile_dim1=0, tile_dim2=0 (1-D)
    g1[5] = 8192;                  // tensor_dim0_stride lo32
    g1[6] = 0;                     // stride hi | tensor_dim1_stride lo
    g1[7] = 0;
    i32x4 gz = {0, 0, 0, 0};
#if __clang_major__ >= 23
    i32x8 gz8 = {0, 0, 0, 0, 0, 0, 0, 0};
    __builtin_amdgcn_tensor_load_to_lds(g0, g1, gz, gz, gz8, 0);
#else
    __builtin_amdgcn_tensor_load_to_lds(g0, g1, gz, gz, 0);
#endif
    __builtin_amdgcn_s_wait_tensorcnt(0);
  }
  __syncthreads();
  const bf16_t* Bsrc = sB;      // ds_load_b128 path
#else
  const bf16_t* Bsrc = Bswz;    // global fallback
#endif

  const bf16_t* arow = A + (long)r * 256;
  v8f c = {};
  #pragma unroll
  for (int ks = 0; ks < 8; ++ks) {
    const int kk = ks * 32;
    v8bf lo = *(const v8bf*)(arow + kk + base);        // K = kk+base .. +7
    v8bf hi = *(const v8bf*)(arow + kk + base + 8);    // K = kk+base+8 .. +15
    v16bf a;
    #pragma unroll
    for (int i = 0; i < 8; ++i) { a[i] = lo[i]; a[8 + i] = hi[i]; }
    v16bf b = *(const v16bf*)(Bsrc + (((wave * 8 + ks) * 32 + lane) * 16));
    c = __builtin_amdgcn_wmma_f32_16x16x32_bf16(false, a, false, b, (short)0, c, false, false);
  }
  const int col   = wave * 16 + m;
  const int rbase = r0 + ((lane < 16) ? 0 : 8);
  if (full) {
    #pragma unroll
    for (int v = 0; v < 8; ++v) h2[(long)(rbase + v) * HID + col] = c[v];
  } else {
    #pragma unroll
    for (int v = 0; v < 8; ++v)
      if (rbase + v < n) h2[(long)(rbase + v) * HID + col] = c[v];
  }
}

// alpha for layer 2: one wave per node, full-wave shfl reduction
__global__ void k_alpha2(const float* __restrict__ h2, const float* __restrict__ a_src,
                         const float* __restrict__ a_dst, float* __restrict__ as,
                         float* __restrict__ ad, int n) {
  const int wave = threadIdx.x >> 5;
  const int lane = threadIdx.x & 31;
  long node = (long)blockIdx.x * 8 + wave;
  if (node >= n) return;
  const float* hp = h2 + node * HID + lane * 2;
  float s = hp[0] * a_src[lane * 2] + hp[1] * a_src[lane * 2 + 1];
  float d = hp[0] * a_dst[lane * 2] + hp[1] * a_dst[lane * 2 + 1];
  #pragma unroll
  for (int msk = 1; msk < 32; msk <<= 1) { s += __shfl_xor(s, msk); d += __shfl_xor(d, msk); }
  if (lane == 0) { as[node] = s; ad[node] = d; }
}

__global__ void k_mxinit2(const float* __restrict__ as, const float* __restrict__ ad,
                          float* __restrict__ mx, int n) {
  long i = (long)blockIdx.x * blockDim.x + threadIdx.x;
  if (i >= n) return;
  mx[i] = lrelu(as[i] + ad[i]);
}

__global__ void k_edgemax2(const int* __restrict__ ei, const float* __restrict__ as,
                           const float* __restrict__ ad, float* __restrict__ mx, int ne) {
  long i = (long)blockIdx.x * blockDim.x + threadIdx.x;
  if (i >= ne) return;
  int s = ei[i], d = ei[ne + i];
  atomicMaxF(&mx[d], lrelu(as[s] + ad[d]));
}

__global__ void k_msg2(const int* __restrict__ ei, const float* __restrict__ h2,
                       const float* __restrict__ as, const float* __restrict__ ad,
                       const float* __restrict__ mx, float* __restrict__ den,
                       float* __restrict__ oacc, int ne, int n) {
  long i = (long)blockIdx.x * blockDim.x + threadIdx.x;
  long total = (long)(ne + n) * HID;
  if (i >= total) return;
  int  c = (int)(i & 63);
  long p = i >> 6;
  int s, d;
  if (p < ne) { s = ei[p]; d = ei[ne + p]; } else { s = d = (int)(p - ne); }
  float ex = expf(lrelu(as[s] + ad[d]) - mx[d]);
  if (c == 0) atomicAdd(&den[d], ex);
  atomicAdd(&oacc[(long)d * HID + c], ex * h2[(long)s * HID + c]);
}

// finalize layer 2 (ELU) + per-graph mean-pool accumulation (channel-parallel)
__global__ void k_pool(const float* __restrict__ oacc, const float* __restrict__ den,
                       const float* __restrict__ b2, const int* __restrict__ batch,
                       float* __restrict__ pooled, float* __restrict__ cnt, int n) {
  long i = (long)blockIdx.x * blockDim.x + threadIdx.x;
  if (i >= (long)n * HID) return;
  int  c    = (int)(i & 63);
  long node = i >> 6;
  int  g    = batch[node];
  if (c == 0) atomicAdd(&cnt[g], 1.f);
  float v = elu(oacc[node * HID + c] / den[node] + b2[c]);
  atomicAdd(&pooled[(long)g * HID + c], v);
}

// mean -> L2 normalize -> linear head
__global__ void k_final(const float* __restrict__ pooled, const float* __restrict__ cnt,
                        const float* __restrict__ Wf, const float* __restrict__ bfp,
                        float* __restrict__ out) {
  int g = threadIdx.x;
  if (g >= NGR) return;
  float c = cnt[g]; c = c > 1.f ? c : 1.f;
  float ss = 0.f, dot = 0.f;
  #pragma unroll 4
  for (int i = 0; i < HID; ++i) {
    float m = pooled[g * HID + i] / c;
    ss += m * m; dot += m * Wf[i];
  }
  float nrm = sqrtf(ss); nrm = nrm > 1e-12f ? nrm : 1e-12f;
  out[g] = dot / nrm + bfp[0];
}

// ---------------------------------------------------------------------------
extern "C" void kernel_launch(void* const* d_in, const int* in_sizes, int n_in,
                              void* d_out, int out_size, void* d_ws, size_t ws_size,
                              hipStream_t stream) {
  const float* x      = (const float*)d_in[0];
  const int*   ei     = (const int*)  d_in[1];
  const int*   batch  = (const int*)  d_in[2];
  const float* W1     = (const float*)d_in[3];
  const float* a_src1 = (const float*)d_in[4];
  const float* a_dst1 = (const float*)d_in[5];
  const float* b1     = (const float*)d_in[6];
  const float* W2     = (const float*)d_in[7];
  const float* a_src2 = (const float*)d_in[8];
  const float* a_dst2 = (const float*)d_in[9];
  const float* b2     = (const float*)d_in[10];
  const float* Wf     = (const float*)d_in[11];
  const float* bfp    = (const float*)d_in[12];
  float* out = (float*)d_out;

  const int N = in_sizes[0] / 4;   // x is [N,4]
  const int E = in_sizes[1] / 2;   // edge_index is [2,E]

  // ---- workspace carve-out (256B aligned slabs) ----
  char* w = (char*)d_ws;
  auto alloc = [&](size_t bytes) -> char* {
    char* p = w; w += (bytes + 255) & ~(size_t)255; return p;
  };
  float*  h1    = (float*) alloc((size_t)N * 256 * 4);   // recycled for h2/o2acc
  float*  o1acc = (float*) alloc((size_t)N * 256 * 4);
  bf16_t* o1bf  = (bf16_t*)alloc((size_t)N * 256 * 2);
  float*  as1   = (float*) alloc((size_t)N * HEADS * 4);
  float*  ad1   = (float*) alloc((size_t)N * HEADS * 4);
  float*  mx1   = (float*) alloc((size_t)N * HEADS * 4);
  float*  den1  = (float*) alloc((size_t)N * HEADS * 4);
  float*  as2   = (float*) alloc((size_t)N * 4);
  float*  ad2   = (float*) alloc((size_t)N * 4);
  float*  mx2   = (float*) alloc((size_t)N * 4);
  float*  den2  = (float*) alloc((size_t)N * 4);
  bf16_t* w2swz = (bf16_t*)alloc((size_t)W2SWZ_ELEMS * 2);
  float*  pooled= (float*) alloc((size_t)NGR * HID * 4);
  float*  cnt   = (float*) alloc((size_t)NGR * 4);
  float* h2    = h1;                       // [N,64]
  float* o2acc = h1 + (size_t)N * HID;     // [N,64]

  const int B = 256;
  auto nb = [&](long t) -> int { return (int)((t + B - 1) / B); };

  // ---- layer 1 ----
  k_zero<<<nb((long)N * 256), B, 0, stream>>>(o1acc, (long)N * 256);
  k_zero<<<nb((long)N * HEADS), B, 0, stream>>>(den1, (long)N * HEADS);
  k_zero<<<nb(NGR * HID), B, 0, stream>>>(pooled, NGR * HID);
  k_zero<<<1, 64, 0, stream>>>(cnt, NGR);

  k_gemm1<<<(N + 15) / 16, 256, 0, stream>>>(x, W1, h1, N);
  k_alpha1<<<(N + 7) / 8, 256, 0, stream>>>(h1, a_src1, a_dst1, as1, ad1, N);
  k_mxinit1<<<nb((long)N * HEADS), B, 0, stream>>>(as1, ad1, mx1, N);
  k_edgemax1<<<nb((long)E * HEADS), B, 0, stream>>>(ei, as1, ad1, mx1, E);
  k_msg1<<<nb((long)(E + N) * HEADS * HID), B, 0, stream>>>(ei, h1, as1, ad1, mx1, den1, o1acc, E, N);
  k_fin1<<<nb((long)N * 256), B, 0, stream>>>(o1acc, den1, b1, o1bf, N);

  // ---- layer 2 ----
  k_zero<<<nb((long)N * HID), B, 0, stream>>>(o2acc, (long)N * HID);
  k_zero<<<nb(N), B, 0, stream>>>(den2, N);
  k_prepw2<<<W2SWZ_ELEMS / 256, 256, 0, stream>>>(W2, w2swz);
  k_gemm2<<<(N + 15) / 16, 128, 0, stream>>>(o1bf, w2swz, h2, N);
  k_alpha2<<<(N + 7) / 8, 256, 0, stream>>>(h2, a_src2, a_dst2, as2, ad2, N);
  k_mxinit2<<<nb(N), B, 0, stream>>>(as2, ad2, mx2, N);
  k_edgemax2<<<nb(E), B, 0, stream>>>(ei, as2, ad2, mx2, E);
  k_msg2<<<nb((long)(E + N) * HID), B, 0, stream>>>(ei, h2, as2, ad2, mx2, den2, o2acc, E, N);

  // ---- pool + head ----
  k_pool<<<nb((long)N * HID), B, 0, stream>>>(o2acc, den2, b2, batch, pooled, cnt, N);
  k_final<<<1, 64, 0, stream>>>(pooled, cnt, Wf, bfp, out);
}
